// TGATUNet_20229295964932
// MI455X (gfx1250) — matrix operand order repaired and checked by reading, not compiled, not run on hardware
//
#include <hip/hip_runtime.h>
#include <math.h>

// ---------------- model constants ----------------
#define T_NODES 4096
#define C_IO    64
#define HID     256
#define HEADS   4
#define DFF     512

typedef __attribute__((ext_vector_type(16))) __bf16 v16bf;
typedef __attribute__((ext_vector_type(8)))  __bf16 v8bf;
typedef __attribute__((ext_vector_type(8)))  float  v8f;

#define WMMA_BF16(a, b, c) \
  __builtin_amdgcn_wmma_f32_16x16x32_bf16(false, (a), false, (b), (short)0, (c), false, false)

__device__ __forceinline__ unsigned pack_bf16(float a, float b) {
  union { __bf16 h[2]; unsigned u; } p;
  p.h[0] = (__bf16)a; p.h[1] = (__bf16)b;
  return p.u;
}

__device__ __forceinline__ v16bf concat8(v8bf lo, v8bf hi) {
  union { v16bf v; v8bf h[2]; } u;
  u.h[0] = lo; u.h[1] = hi;
  return u.v;
}

// A fragment (16x32 bf16) straight from global f32, using 16B vector loads.
// lane L: row = row0 + (L&15); lanes>=16 -> K+8; element j -> K = (j&7) + (j&8?16:0).
__device__ __forceinline__ v16bf load_a_frag_f4(const float* A, int lda, int row0, int k0) {
  const int lane = threadIdx.x & 31;
  const float* p = A + (size_t)(row0 + (lane & 15)) * lda + k0 + ((lane & 16) ? 8 : 0);
  const float4 x0 = *(const float4*)(p);
  const float4 x1 = *(const float4*)(p + 4);
  const float4 x2 = *(const float4*)(p + 16);
  const float4 x3 = *(const float4*)(p + 20);
  v16bf a;
  a[0]=(__bf16)x0.x; a[1]=(__bf16)x0.y; a[2]=(__bf16)x0.z; a[3]=(__bf16)x0.w;
  a[4]=(__bf16)x1.x; a[5]=(__bf16)x1.y; a[6]=(__bf16)x1.z; a[7]=(__bf16)x1.w;
  a[8]=(__bf16)x2.x; a[9]=(__bf16)x2.y; a[10]=(__bf16)x2.z; a[11]=(__bf16)x2.w;
  a[12]=(__bf16)x3.x; a[13]=(__bf16)x3.y; a[14]=(__bf16)x3.z; a[15]=(__bf16)x3.w;
  return a;
}

// =====================================================================
// GEMM: C[N x M] = act(A[N x K] @ B[K x M] + bias)
// block 128 (4 waves) -> 64x64 tile. Double-buffered LDS bf16 tiles,
// B stored transposed so every fragment read is two contiguous 16B ds_loads.
// Next-tile global loads issued before the WMMAs (latency hiding).
// grid (M/64, N/64).
// =====================================================================
__global__ void gemm_wmma(const float* __restrict__ A, const float* __restrict__ B,
                          const float* __restrict__ bias, float* __restrict__ C,
                          int N, int M, int K, int act) {
  __shared__ __align__(16) __bf16 ldsA[2][64 * 32];   // [row][k]
  __shared__ __align__(16) __bf16 ldsB[2][64 * 32];   // [col][k] (transposed)

  const int t = threadIdx.x;          // 0..127
  const int wave = t >> 5;
  const int lane = t & 31;
  const int col0 = blockIdx.x * 64;
  const int rowB0 = blockIdx.y * 64;
  const int row0 = rowB0 + wave * 16;

  v8f acc[4] = {v8f{0}, v8f{0}, v8f{0}, v8f{0}};

  // ---- prologue: stage k-tile 0 into buffer 0 (uniform 4-step loops) ----
  {
    unsigned* uA = (unsigned*)ldsA[0];
#pragma unroll
    for (int it = 0; it < 4; ++it) {
      const int i = t + 128 * it;
      const int r = i >> 3, c4 = i & 7;                 // 8 float4 per A row
      const float4 f = *(const float4*)(A + (size_t)(rowB0 + r) * K + c4 * 4);
      uA[r * 16 + c4 * 2]     = pack_bf16(f.x, f.y);
      uA[r * 16 + c4 * 2 + 1] = pack_bf16(f.z, f.w);
    }
#pragma unroll
    for (int it = 0; it < 4; ++it) {
      const int i = t + 128 * it;
      const int k = i >> 4, c4 = i & 15;                // 16 float4 per B k-row
      const float4 f = *(const float4*)(B + (size_t)k * M + col0 + c4 * 4);
      const int n = c4 * 4;
      ldsB[0][(n + 0) * 32 + k] = (__bf16)f.x;
      ldsB[0][(n + 1) * 32 + k] = (__bf16)f.y;
      ldsB[0][(n + 2) * 32 + k] = (__bf16)f.z;
      ldsB[0][(n + 3) * 32 + k] = (__bf16)f.w;
    }
  }

  int p = 0;
  for (int k0 = 0; k0 < K; k0 += 32) {
    __syncthreads();                                    // buffer p ready
    const int kn = k0 + 32;
    const bool next = kn < K;

    // ---- issue next-tile global loads early (hidden behind WMMAs) ----
    float4 fa[4], fb[4];
    if (next) {
#pragma unroll
      for (int it = 0; it < 4; ++it) {
        const int i = t + 128 * it;
        fa[it] = *(const float4*)(A + (size_t)(rowB0 + (i >> 3)) * K + kn + (i & 7) * 4);
        fb[it] = *(const float4*)(B + (size_t)(kn + (i >> 4)) * M + col0 + (i & 15) * 4);
      }
    }

    // ---- compute from buffer p ----
    const __bf16* lA = ldsA[p];
    const __bf16* lB = ldsB[p];
    const int ml = wave * 16 + (lane & 15);
    const int kha = (lane & 16) ? 8 : 0;
    const v16bf a = concat8(*(const v8bf*)&lA[ml * 32 + kha],
                            *(const v8bf*)&lA[ml * 32 + kha + 16]);
    const int khb = (lane & 16) ? 8 : 0;
    v16bf bfr[4];
#pragma unroll
    for (int tt = 0; tt < 4; ++tt)
      bfr[tt] = concat8(*(const v8bf*)&lB[(tt * 16 + (lane & 15)) * 32 + khb],
                        *(const v8bf*)&lB[(tt * 16 + (lane & 15)) * 32 + khb + 16]);
#pragma unroll
    for (int tt = 0; tt < 4; ++tt)
      acc[tt] = WMMA_BF16(a, bfr[tt], acc[tt]);

    // ---- write next tile into buffer p^1 ----
    if (next) {
      unsigned* uA = (unsigned*)ldsA[p ^ 1];
      __bf16* vB = ldsB[p ^ 1];
#pragma unroll
      for (int it = 0; it < 4; ++it) {
        const int i = t + 128 * it;
        {
          const int r = i >> 3, c4 = i & 7;
          uA[r * 16 + c4 * 2]     = pack_bf16(fa[it].x, fa[it].y);
          uA[r * 16 + c4 * 2 + 1] = pack_bf16(fa[it].z, fa[it].w);
        }
        {
          const int k = i >> 4, c4 = i & 15;
          const int n = c4 * 4;
          vB[(n + 0) * 32 + k] = (__bf16)fb[it].x;
          vB[(n + 1) * 32 + k] = (__bf16)fb[it].y;
          vB[(n + 2) * 32 + k] = (__bf16)fb[it].z;
          vB[(n + 3) * 32 + k] = (__bf16)fb[it].w;
        }
      }
    }
    p ^= 1;
  }

  // ---- epilogue ----
  const int n = lane & 15;
  const int mh = (lane & 16) ? 8 : 0;
#pragma unroll
  for (int tt = 0; tt < 4; ++tt) {
    const int col = col0 + 16 * tt + n;
    const float bv = bias ? bias[col] : 0.0f;
#pragma unroll
    for (int r = 0; r < 8; ++r) {
      float v = acc[tt][r] + bv;
      if (act) v = fmaxf(v, 0.0f);
      C[(size_t)(row0 + mh + r) * M + col] = v;
    }
  }
}

// =====================================================================
// Flash attention: O = softmax(Q K^T / 8) V.
// block 128 (4 waves, 4 query blocks of one head) shares double-buffered
// LDS K [key][dim] / transposed V [dim][key] tiles (4x HBM reuse).
// S tile lives in WMMA C/D registers; P re-laid-out via per-wave LDS.
// grid (T/64, HEADS).
// =====================================================================
__global__ void attn_flash(const float* __restrict__ Q, const float* __restrict__ Km,
                           const float* __restrict__ V, float* __restrict__ O, int T) {
  __shared__ __align__(16) __bf16 ldsK[2][32 * 64];    // [key][dim]
  __shared__ __align__(16) __bf16 ldsV[2][64 * 32];    // [dim][key] (transposed)
  __shared__ __align__(16) __bf16 ldsP[4 * 16 * 32];   // per-wave P tile

  const int t = threadIdx.x;
  const int wave = t >> 5;
  const int lane = t & 31;
  const int half = (lane & 16) ? 1 : 0;
  const int row0 = blockIdx.x * 64 + wave * 16;
  const int cbase = blockIdx.y * 64;
  const float scale = 0.125f;                          // 1/sqrt(64)
  __bf16* myP = ldsP + wave * (16 * 32);

  const v16bf aq0 = load_a_frag_f4(Q + cbase, HID, row0, 0);
  const v16bf aq1 = load_a_frag_f4(Q + cbase, HID, row0, 32);

  v8f o0 = v8f{0}, o1 = v8f{0}, o2 = v8f{0}, o3 = v8f{0};
  float mrun[8], lrun[8];
#pragma unroll
  for (int r = 0; r < 8; ++r) { mrun[r] = -1e30f; lrun[r] = 0.0f; }

  // ---- prologue: stage key-block 0 into buffer 0 ----
  {
    unsigned* uK = (unsigned*)ldsK[0];
#pragma unroll
    for (int it = 0; it < 4; ++it) {
      const int i = t + 128 * it;
      const int key = i >> 4, c4 = i & 15;
      const float4 fk = *(const float4*)(Km + (size_t)key * HID + cbase + c4 * 4);
      uK[key * 32 + c4 * 2]     = pack_bf16(fk.x, fk.y);
      uK[key * 32 + c4 * 2 + 1] = pack_bf16(fk.z, fk.w);
      const float4 fv = *(const float4*)(V + (size_t)key * HID + cbase + c4 * 4);
      const int n = c4 * 4;
      ldsV[0][(n + 0) * 32 + key] = (__bf16)fv.x;
      ldsV[0][(n + 1) * 32 + key] = (__bf16)fv.y;
      ldsV[0][(n + 2) * 32 + key] = (__bf16)fv.z;
      ldsV[0][(n + 3) * 32 + key] = (__bf16)fv.w;
    }
  }

  int p = 0;
  for (int kb = 0; kb < T; kb += 32) {
    __syncthreads();                                   // buffer p ready
    const int kn = kb + 32;
    const bool next = kn < T;

    // ---- issue next key-block global loads early ----
    float4 fk[4], fv[4];
    if (next) {
#pragma unroll
      for (int it = 0; it < 4; ++it) {
        const int i = t + 128 * it;
        const int key = kn + (i >> 4), c4 = i & 15;
        fk[it] = *(const float4*)(Km + (size_t)key * HID + cbase + c4 * 4);
        fv[it] = *(const float4*)(V + (size_t)key * HID + cbase + c4 * 4);
      }
    }

    // ---- S = Q @ K^T (two 16x16 D tiles) ----
    const __bf16* lK = ldsK[p];
    const __bf16* lV = ldsV[p];
    const int khb = half ? 8 : 0;
    const int keyA = (lane & 15);
    const v16bf k00 = concat8(*(const v8bf*)&lK[keyA * 64 + khb], *(const v8bf*)&lK[keyA * 64 + khb + 16]);
    const v16bf k01 = concat8(*(const v8bf*)&lK[keyA * 64 + 32 + khb], *(const v8bf*)&lK[keyA * 64 + 32 + khb + 16]);
    const v16bf k10 = concat8(*(const v8bf*)&lK[(16 + keyA) * 64 + khb], *(const v8bf*)&lK[(16 + keyA) * 64 + khb + 16]);
    const v16bf k11 = concat8(*(const v8bf*)&lK[(16 + keyA) * 64 + 32 + khb], *(const v8bf*)&lK[(16 + keyA) * 64 + 32 + khb + 16]);
    v8f s0 = v8f{0}, s1 = v8f{0};
    s0 = WMMA_BF16(aq0, k00, s0);
    s0 = WMMA_BF16(aq1, k01, s0);
    s1 = WMMA_BF16(aq0, k10, s1);
    s1 = WMMA_BF16(aq1, k11, s1);

    // ---- online softmax: rows live in 16-lane halves of the C/D layout ----
#pragma unroll
    for (int r = 0; r < 8; ++r) {
      const float e0 = s0[r] * scale;
      const float e1 = s1[r] * scale;
      float rmax = fmaxf(e0, e1);
#pragma unroll
      for (int m = 1; m < 16; m <<= 1) rmax = fmaxf(rmax, __shfl_xor(rmax, m, 32));
      const float mnew = fmaxf(mrun[r], rmax);
      const float alpha = __expf(mrun[r] - mnew);
      o0[r] *= alpha; o1[r] *= alpha; o2[r] *= alpha; o3[r] *= alpha;
      const float p0 = __expf(e0 - mnew);
      const float p1 = __expf(e1 - mnew);
      float rs = p0 + p1;
#pragma unroll
      for (int m = 1; m < 16; m <<= 1) rs += __shfl_xor(rs, m, 32);
      lrun[r] = lrun[r] * alpha + rs;
      mrun[r] = mnew;
      const int row = r + half * 8;
      myP[row * 32 + (lane & 15)]      = (__bf16)p0;
      myP[row * 32 + 16 + (lane & 15)] = (__bf16)p1;
    }
    // per-wave LDS turnaround: DS pipeline is in-order per wave; the wait
    // (+ memory clobber) keeps the compiler from hoisting the reads.
    // (Issued BEFORE the next-tile ds_stores so it never waits on staging.)
    asm volatile("s_wait_dscnt 0" ::: "memory");

    // ---- re-read P as an A fragment (16x32) ----
    const int mp = lane & 15;
    const int khp = half * 8;
    const v16bf pa = concat8(*(const v8bf*)&myP[mp * 32 + khp],
                             *(const v8bf*)&myP[mp * 32 + khp + 16]);

    // ---- O += P @ V ----
    const v16bf vf0 = concat8(*(const v8bf*)&lV[(0  + keyA) * 32 + khb], *(const v8bf*)&lV[(0  + keyA) * 32 + khb + 16]);
    const v16bf vf1 = concat8(*(const v8bf*)&lV[(16 + keyA) * 32 + khb], *(const v8bf*)&lV[(16 + keyA) * 32 + khb + 16]);
    const v16bf vf2 = concat8(*(const v8bf*)&lV[(32 + keyA) * 32 + khb], *(const v8bf*)&lV[(32 + keyA) * 32 + khb + 16]);
    const v16bf vf3 = concat8(*(const v8bf*)&lV[(48 + keyA) * 32 + khb], *(const v8bf*)&lV[(48 + keyA) * 32 + khb + 16]);
    o0 = WMMA_BF16(pa, vf0, o0);
    o1 = WMMA_BF16(pa, vf1, o1);
    o2 = WMMA_BF16(pa, vf2, o2);
    o3 = WMMA_BF16(pa, vf3, o3);

    // ---- write next key-block into buffer p^1 ----
    if (next) {
      unsigned* uK = (unsigned*)ldsK[p ^ 1];
      __bf16* vB = ldsV[p ^ 1];
#pragma unroll
      for (int it = 0; it < 4; ++it) {
        const int i = t + 128 * it;
        const int key = i >> 4, c4 = i & 15;
        uK[key * 32 + c4 * 2]     = pack_bf16(fk[it].x, fk[it].y);
        uK[key * 32 + c4 * 2 + 1] = pack_bf16(fk[it].z, fk[it].w);
        const int n = c4 * 4;
        vB[(n + 0) * 32 + key] = (__bf16)fv[it].x;
        vB[(n + 1) * 32 + key] = (__bf16)fv[it].y;
        vB[(n + 2) * 32 + key] = (__bf16)fv[it].z;
        vB[(n + 3) * 32 + key] = (__bf16)fv[it].w;
      }
    }
    p ^= 1;
  }

  // ---- epilogue ----
  const int n = lane & 15;
#pragma unroll
  for (int r = 0; r < 8; ++r) {
    const float inv = 1.0f / lrun[r];
    const size_t row = (size_t)(row0 + half * 8 + r) * HID + cbase;
    O[row + 0  + n] = o0[r] * inv;
    O[row + 16 + n] = o1[r] * inv;
    O[row + 32 + n] = o2[r] * inv;
    O[row + 48 + n] = o3[r] * inv;
  }
}

// =====================================================================
// LayerNorm(x + y) * g + b ; one wave per row, D = 256 (8 elems/lane)
// =====================================================================
__global__ void ln_residual(const float* __restrict__ x, const float* __restrict__ y,
                            const float* __restrict__ g, const float* __restrict__ b,
                            float* __restrict__ out, int D) {
  const int row = blockIdx.x;
  const int lane = threadIdx.x;
  float v[8];
  float s = 0.0f;
#pragma unroll
  for (int i = 0; i < 8; ++i) {
    const int c = lane + 32 * i;
    v[i] = x[(size_t)row * D + c] + y[(size_t)row * D + c];
    s += v[i];
  }
#pragma unroll
  for (int m = 1; m < 32; m <<= 1) s += __shfl_xor(s, m, 32);
  const float mean = s / (float)D;
  float ss = 0.0f;
#pragma unroll
  for (int i = 0; i < 8; ++i) { const float d = v[i] - mean; ss += d * d; }
#pragma unroll
  for (int m = 1; m < 32; m <<= 1) ss += __shfl_xor(ss, m, 32);
  const float rstd = rsqrtf(ss / (float)D + 1e-5f);
#pragma unroll
  for (int i = 0; i < 8; ++i) {
    const int c = lane + 32 * i;
    out[(size_t)row * D + c] = (v[i] - mean) * rstd * g[c] + b[c];
  }
}

// =====================================================================
// GAT helpers. Edge list is the fixed +-1..4 hop stencil + self loops,
// so attention is a 9-tap per-node softmax (no atomics, no segment ops).
// =====================================================================
__global__ void gat_scores(const float* __restrict__ h, const float* __restrict__ a_s,
                           const float* __restrict__ a_d, float* __restrict__ ss,
                           float* __restrict__ sd, int N, int H, int F) {
  const int i = blockIdx.x * blockDim.x + threadIdx.x;
  if (i >= N * H) return;
  const int n = i / H, hh = i % H;
  const float* hp = h + (size_t)n * H * F + hh * F;
  float as = 0.0f, ad = 0.0f;
  for (int f = 0; f < F; ++f) {
    as += hp[f] * a_s[hh * F + f];
    ad += hp[f] * a_d[hh * F + f];
  }
  ss[i] = as;
  sd[i] = ad;
}

__global__ void gat_att(const float* __restrict__ ss, const float* __restrict__ sd,
                        float* __restrict__ att, int N, int H) {
  const int i = blockIdx.x * blockDim.x + threadIdx.x;
  if (i >= N * H) return;
  const int n = i / H, hh = i % H;
  const float sdv = sd[i];
  float e[9];
  float mx = -1e30f;
#pragma unroll
  for (int j = 0; j < 9; ++j) {
    const int src = n + (j - 4);
    if (src >= 0 && src < N) {
      float tv = ss[src * H + hh] + sdv;
      tv = (tv > 0.0f) ? tv : 0.2f * tv;    // leaky_relu(.,0.2)
      e[j] = tv;
      mx = fmaxf(mx, tv);
    } else {
      e[j] = -1e30f;
    }
  }
  float denom = 0.0f;
  float ex[9];
#pragma unroll
  for (int j = 0; j < 9; ++j) {
    const int src = n + (j - 4);
    ex[j] = (src >= 0 && src < N) ? __expf(e[j] - mx) : 0.0f;
    denom += ex[j];
  }
  const float inv = 1.0f / denom;
#pragma unroll
  for (int j = 0; j < 9; ++j) att[(size_t)i * 9 + j] = ex[j] * inv;
}

__global__ void gat_aggr(const float* __restrict__ h, const float* __restrict__ att,
                         const float* __restrict__ bias, float* __restrict__ out,
                         int N, int H, int F, int act, int transposed) {
  const int HF = H * F;
  const int i = blockIdx.x * blockDim.x + threadIdx.x;
  if (i >= N * HF) return;
  const int n = i / HF, c = i % HF;
  const int hh = c / F;
  const float* ap = att + ((size_t)n * H + hh) * 9;
  float acc = bias[c];
#pragma unroll
  for (int j = 0; j < 9; ++j) {
    const int src = n + (j - 4);
    if (src >= 0 && src < N) acc += ap[j] * h[(size_t)src * HF + c];
  }
  if (act) acc = fmaxf(acc, 0.0f);
  if (transposed) out[(size_t)c * N + n] = acc;   // final layer writes y.T [C,T]
  else            out[i] = acc;
}

// =====================================================================
// classifier head
// =====================================================================
__global__ void col_mean(const float* __restrict__ x, float* __restrict__ out, int N, int D) {
  const int c = blockIdx.x * blockDim.x + threadIdx.x;
  if (c >= D) return;
  float s = 0.0f;
  for (int n = 0; n < N; ++n) s += x[(size_t)n * D + c];
  out[c] = s / (float)N;
}

__global__ void classifier(const float* __restrict__ hcls, const float* __restrict__ W1,
                           const float* __restrict__ b1, const float* __restrict__ W2,
                           const float* __restrict__ b2, float* __restrict__ logits) {
  __shared__ float hid[128];
  const int t = threadIdx.x;
  float s = b1[t];
  for (int k = 0; k < HID; ++k) s += hcls[k] * W1[k * 128 + t];
  hid[t] = fmaxf(s, 0.0f);
  __syncthreads();
  if (t < 2) {
    float s2 = b2[t];
    for (int k = 0; k < 128; ++k) s2 += hid[k] * W2[k * 2 + t];
    logits[t] = s2;
  }
}

// =====================================================================
// host orchestration
// =====================================================================
// Input order: [0]=window, [1]=edge_index (unused: fixed stencil graph),
// then params flattened with sorted dict keys (JAX pytree order):
//   bott[0]: W1 W2 Wk Wo Wq Wv b1n b2n bf1 bf2 bk bo bq bv g1 g2   (2..17)
//   bott[1]: same                                                  (18..33)
//   cls:     W1 W2 b1 b2                                           (34..37)
//   dec[0..2]: W a_d a_s b                                         (38..49)
//   enc[0..2]: W a_d a_s b                                         (50..61)
extern "C" void kernel_launch(void* const* d_in, const int* in_sizes, int n_in,
                              void* d_out, int out_size, void* d_ws, size_t ws_size,
                              hipStream_t stream) {
  (void)in_sizes; (void)n_in; (void)out_size; (void)ws_size;
  const int T = T_NODES;
  const float* window = (const float*)d_in[0];
  float* out = (float*)d_out;

  auto P = [&](int i) { return (const float*)d_in[i]; };
  const int BOTT0 = 2, BOTT1 = 18;
  const int B_W1 = 0, B_W2 = 1, B_Wk = 2, B_Wo = 3, B_Wq = 4, B_Wv = 5,
            B_b1n = 6, B_b2n = 7, B_bf1 = 8, B_bf2 = 9, B_bk = 10, B_bo = 11,
            B_bq = 12, B_bv = 13, B_g1 = 14, B_g2 = 15;
  const int CLS_W1 = 34, CLS_W2 = 35, CLS_b1 = 36, CLS_b2 = 37;
  const int DEC0 = 38, DEC1 = 42, DEC2 = 46;
  const int ENC0 = 50, ENC1 = 54, ENC2 = 58;

  // ---- workspace carve (floats), ~39 MB ----
  float* w = (float*)d_ws;
  const size_t NH = (size_t)T * HID;
  float* xa   = w;               w += NH;            // x ping
  float* xb   = w;               w += NH;            // x pong
  float* hbuf = w;               w += NH;            // GAT h / LN temp
  float* q    = w;               w += NH;
  float* kk   = w;               w += NH;
  float* v    = w;               w += NH;
  float* o    = w;               w += NH;
  float* ff   = w;               w += (size_t)T * DFF;
  float* ss   = w;               w += (size_t)T * HEADS;
  float* sd   = w;               w += (size_t)T * HEADS;
  float* att  = w;               w += (size_t)T * HEADS * 9;
  float* hcls = w;               w += HID;

  auto gemm = [&](const float* A, const float* B, const float* bias, float* Cm,
                  int N, int M, int K, int act) {
    gemm_wmma<<<dim3(M / 64, N / 64), 128, 0, stream>>>(A, B, bias, Cm, N, M, K, act);
  };

  auto gat = [&](const float* xin, int Kin, int pi, int H, int F, float* xout,
                 int act, int transposed) {
    const int M = H * F;
    gemm(xin, P(pi + 0), nullptr, hbuf, T, M, Kin, 0);              // h = x @ W
    const int NHth = T * H;
    gat_scores<<<(NHth + 255) / 256, 256, 0, stream>>>(hbuf, P(pi + 2), P(pi + 1), ss, sd, T, H, F);
    gat_att<<<(NHth + 255) / 256, 256, 0, stream>>>(ss, sd, att, T, H);
    gat_aggr<<<(T * M + 255) / 256, 256, 0, stream>>>(hbuf, att, P(pi + 3), xout, T, H, F, act, transposed);
  };

  auto tlayer = [&](float* x, int b) {   // x updated in place (xa)
    gemm(x, P(b + B_Wq), P(b + B_bq), q, T, HID, HID, 0);
    gemm(x, P(b + B_Wk), P(b + B_bk), kk, T, HID, HID, 0);
    gemm(x, P(b + B_Wv), P(b + B_bv), v, T, HID, HID, 0);
    attn_flash<<<dim3(T / 64, HEADS), 128, 0, stream>>>(q, kk, v, o, T);
    gemm(o, P(b + B_Wo), P(b + B_bo), xb, T, HID, HID, 0);
    ln_residual<<<T, 32, 0, stream>>>(x, xb, P(b + B_g1), P(b + B_b1n), hbuf, HID);
    gemm(hbuf, P(b + B_W1), P(b + B_bf1), ff, T, DFF, HID, 1);      // relu
    gemm(ff, P(b + B_W2), P(b + B_bf2), xb, T, HID, DFF, 0);
    ln_residual<<<T, 32, 0, stream>>>(hbuf, xb, P(b + B_g2), P(b + B_b2n), x, HID);
  };

  // ---- encoder ----
  gat(window, C_IO, ENC0, HEADS, HID / HEADS, xa, 1, 0);
  gat(xa,     HID,  ENC1, HEADS, HID / HEADS, xb, 1, 0);
  gat(xb,     HID,  ENC2, HEADS, HID / HEADS, xa, 1, 0);

  // ---- bottleneck transformers ----
  tlayer(xa, BOTT0);
  tlayer(xa, BOTT1);

  // ---- classifier (logits go after y.T in d_out) ----
  col_mean<<<1, 256, 0, stream>>>(xa, hcls, T, HID);
  classifier<<<1, 128, 0, stream>>>(hcls, P(CLS_W1), P(CLS_b1), P(CLS_W2), P(CLS_b2),
                                    out + (size_t)C_IO * T);

  // ---- decoder ----
  gat(xa, HID, DEC0, HEADS, HID / HEADS, xb, 1, 0);
  gat(xb, HID, DEC1, HEADS, HID / HEADS, xa, 1, 0);
  gat(xa, HID, DEC2, 1, C_IO, out, 0, 1);   // final: H=1, F=64, no relu, write y.T
}
